// MoE_13537736917214
// MI455X (gfx1250) — matrix operand order repaired and checked by reading, not compiled, not run on hardware
//
#include <hip/hip_runtime.h>
#include <math.h>

typedef __attribute__((ext_vector_type(16))) _Float16 v16h;
typedef __attribute__((ext_vector_type(8)))  _Float16 v8h;
typedef __attribute__((ext_vector_type(8)))  float    v8f;

#define FIVE_PI 15.7079632679489662f

// packed-weight sizes (in halves)
#define EWM_PACK (16*3*128*128)   // 786432
#define GWM_PACK (2*64*64)        // 8192
#define GWO_PACK (64*16)          // 1024

// ---------------------------------------------------------------------------
// Branch-free tanh: sign(x) * (1-e)/(1+e), e = exp2(-2*log2(e)*|x|) in (0,1].
// Lowers to v_exp_f32 + v_rcp_f32 (TRANS) + a few VALU; no exec-mask branches.
// ---------------------------------------------------------------------------
__device__ __forceinline__ float fast_tanh(float x) {
    float a = __builtin_fabsf(x);
    float e = __builtin_amdgcn_exp2f(a * -2.88539008177792681f); // exp(-2a)
    float r = (1.0f - e) * __builtin_amdgcn_rcpf(1.0f + e);
    return __builtin_copysignf(r, x);
}

// ---------------------------------------------------------------------------
// WMMA wrapper: D = A(16x32 f16) * B(32x16 f16) + C(16x16 f32)
// ---------------------------------------------------------------------------
__device__ __forceinline__ v8f wmma_f16(v16h a, v16h b, v8f c) {
    return __builtin_amdgcn_wmma_f32_16x16x32_f16(
        /*neg_a=*/false, a, /*neg_b=*/false, b,
        /*c_mod=*/(short)0, c, /*reuse_a=*/false, /*reuse_b=*/false);
}

// ---------------------------------------------------------------------------
// A fragment (16x32, f16) from a row-major LDS activation tile.
// ISA 7.12.2: lanes 0-15 hold M=lane, K=k0+0..7 then k0+16..23;
//             lanes 16-31 hold M=lane-16, K=k0+8..15 then k0+24..31.
// Per lane: two groups of 8 contiguous halves -> 2x ds_load_b128.
// ---------------------------------------------------------------------------
__device__ __forceinline__ v16h load_a_lds(const _Float16* act, int ld,
                                           int mb, int k0, int lane) {
    int m  = lane & 15;
    int hf = lane >> 4;
    const _Float16* p = act + (mb + m) * ld + k0 + hf * 8;
    v8h lo = *(const v8h*)(p);
    v8h hi = *(const v8h*)(p + 16);
    v16h r;
#pragma unroll
    for (int i = 0; i < 8; ++i) { r[i] = lo[i]; r[i + 8] = hi[i]; }
    return r;
}

// ---------------------------------------------------------------------------
// B fragment (32x16, f16) from pre-packed global weights.
// Packed layout per k-block: [half(0..1)][n(0..ncols-1)][k16(0..15)],
// so each lane reads 16 contiguous halves (32B) -> 2x global_load_b128.
// ---------------------------------------------------------------------------
__device__ __forceinline__ v16h load_b_pack(const _Float16* wkb, int ncols,
                                            int nb, int lane) {
    int hf = lane >> 4;
    int n  = nb + (lane & 15);
    const _Float16* p = wkb + (hf * ncols + n) * 16;
    v8h lo = *(const v8h*)(p);
    v8h hi = *(const v8h*)(p + 8);
    v16h r;
#pragma unroll
    for (int i = 0; i < 8; ++i) { r[i] = lo[i]; r[i + 8] = hi[i]; }
    return r;
}

// ---------------------------------------------------------------------------
// Epilogue: C (16x16 f32) -> bias + tanh -> f16 row-major LDS tile.
// C layout: VGPR r holds M=r (lanes 0-15) / M=8+r (lanes 16-31), N=lane&15.
// ---------------------------------------------------------------------------
__device__ __forceinline__ void store_act(_Float16* nxt, int ld, int mb,
                                          int nb, v8f c, float bias, int lane) {
    int col = nb + (lane & 15);
    int rb  = mb + ((lane >> 4) << 3);
#pragma unroll
    for (int r = 0; r < 8; ++r)
        nxt[(rb + r) * ld + col] = (_Float16)fast_tanh(c[r] + bias);
}

// ---------------------------------------------------------------------------
// Pre-pass: convert fp32 weights to f16 in WMMA B-fragment order.
// ---------------------------------------------------------------------------
__global__ __launch_bounds__(256) void MoE_pack_kernel(
    const float* __restrict__ ewm, const float* __restrict__ gwm,
    const float* __restrict__ gwo, _Float16* __restrict__ ws) {
    int idx = blockIdx.x * 256 + threadIdx.x;
    if (idx < EWM_PACK) {
        // ewm [e][l][k][n] : el = e*3+l
        int n = idx & 127, k = (idx >> 7) & 127, el = idx >> 14;
        int kb = k >> 5, hf = (k >> 3) & 1, kk = (k & 7) + (((k >> 4) & 1) << 3);
        ws[(((el * 4 + kb) * 2 + hf) * 128 + n) * 16 + kk] = (_Float16)ewm[idx];
    } else if (idx < EWM_PACK + GWM_PACK) {
        int j = idx - EWM_PACK;               // gwm [l][k][n], GH=64
        int n = j & 63, k = (j >> 6) & 63, l = j >> 12;
        int kb = k >> 5, hf = (k >> 3) & 1, kk = (k & 7) + (((k >> 4) & 1) << 3);
        ws[EWM_PACK + (((l * 2 + kb) * 2 + hf) * 64 + n) * 16 + kk] =
            (_Float16)gwm[j];
    } else if (idx < EWM_PACK + GWM_PACK + GWO_PACK) {
        int j = idx - EWM_PACK - GWM_PACK;    // gwo [k][n], 64x16
        int n = j & 15, k = j >> 4;
        int kb = k >> 5, hf = (k >> 3) & 1, kk = (k & 7) + (((k >> 4) & 1) << 3);
        ws[EWM_PACK + GWM_PACK + ((kb * 2 + hf) * 16 + n) * 16 + kk] =
            (_Float16)gwo[j];
    }
}

// ---------------------------------------------------------------------------
// Fused MoE: 64 tokens per workgroup, 8 waves (wave32).
// ---------------------------------------------------------------------------
__global__ __launch_bounds__(256) void MoE_main_kernel(
    const float* __restrict__ x,
    const float* __restrict__ ew1, const float* __restrict__ eb1,
    const float* __restrict__ ebm,
    const float* __restrict__ ewo, const float* __restrict__ ebo,
    const float* __restrict__ gw1, const float* __restrict__ gb1,
    const float* __restrict__ gbm, const float* __restrict__ gbo,
    const _Float16* __restrict__ wmid,   // packed ewm
    const _Float16* __restrict__ wgm,    // packed gwm
    const _Float16* __restrict__ wgo,    // packed gwo
    float* __restrict__ out) {

    __shared__ alignas(16) _Float16 bufA[64 * 128];
    __shared__ alignas(16) _Float16 bufB[64 * 128];
    __shared__ float gates[64 * 16];
    __shared__ float xs[64 * 2];

    const int tid  = threadIdx.x;
    const int wave = tid >> 5;
    const int lane = tid & 31;
    const int n0   = blockIdx.x * 64;

    if (tid < 128) xs[tid] = x[n0 * 2 + tid];
    __syncthreads();

    // ===================== gating network =====================
    // layer 1: [64,2] @ [2,64] (K=2 -> VALU), f16 into bufA (ld=64)
    for (int idx = tid; idx < 64 * 64; idx += 256) {
        int t = idx >> 6, c = idx & 63;
        float v = xs[t * 2] * gw1[c] + xs[t * 2 + 1] * gw1[64 + c] + gb1[c];
        bufA[t * 64 + c] = (_Float16)fast_tanh(v);
    }
    __syncthreads();

    // two 64x64x64 WMMA layers; wave -> (m-pair, n-tile)
    {
        const int nb  = (wave & 3) * 16;
        const int mb0 = (wave >> 2) * 32;
#pragma unroll
        for (int l = 0; l < 2; ++l) {
            const _Float16* cur = (l & 1) ? bufB : bufA;
            _Float16*       nxt = (l & 1) ? bufA : bufB;
            const _Float16* wl  = wgm + l * (2 * 2 * 64 * 16);
            v16h b0 = load_b_pack(wl, 64, nb, lane);                 // kb=0
            v16h b1 = load_b_pack(wl + 2 * 64 * 16, 64, nb, lane);   // kb=32
            float bias = gbm[l * 64 + nb + (lane & 15)];
#pragma unroll
            for (int mi = 0; mi < 2; ++mi) {
                int mb = mb0 + mi * 16;
                v8f acc = {};
                acc = wmma_f16(load_a_lds(cur, 64, mb, 0,  lane), b0, acc);
                acc = wmma_f16(load_a_lds(cur, 64, mb, 32, lane), b1, acc);
                store_act(nxt, 64, mb, nb, acc, bias, lane);
            }
            __syncthreads();
        }
    }

    // gate layer: [64,64] @ [64,16] -> sin(5*pi*z). Result g is in bufA.
    if (wave < 4) {
        const int mb = wave * 16;
        v16h b0 = load_b_pack(wgo, 16, 0, lane);
        v16h b1 = load_b_pack(wgo + 2 * 16 * 16, 16, 0, lane);
        float bias = gbo[lane & 15];
        v8f acc = {};
        acc = wmma_f16(load_a_lds(bufA, 64, mb, 0,  lane), b0, acc);
        acc = wmma_f16(load_a_lds(bufA, 64, mb, 32, lane), b1, acc);
        int col = lane & 15;
        int rb  = mb + ((lane >> 4) << 3);
#pragma unroll
        for (int r = 0; r < 8; ++r)
            gates[(rb + r) * 16 + col] = __sinf(FIVE_PI * (acc[r] + bias));
    }
    __syncthreads();

    // ===================== expert stack =====================
    // Thread 4t+p handles quarter p of token t in the output projection;
    // yacc is carried in threads with p==0.
    const int tok  = tid >> 2;
    const int part = tid & 3;
    float yacc = 0.0f;

    for (int e = 0; e < 16; ++e) {
        // layer 1: [64,2] @ [2,128] (K=2 -> VALU), f16 into bufA (ld=128).
        // Column (idx & 127) is fixed per thread -> weights hoist out of loop.
        for (int idx = tid; idx < 64 * 128; idx += 256) {
            int t = idx >> 7, c = idx & 127;
            float v = xs[t * 2] * ew1[e * 256 + c] +
                      xs[t * 2 + 1] * ew1[e * 256 + 128 + c] +
                      eb1[e * 128 + c];
            bufA[t * 128 + c] = (_Float16)fast_tanh(v);
        }
        __syncthreads();

        // three 64x128x128 WMMA layers; wave owns 16 output columns,
        // B fragments held in registers and reused across 4 M-tiles.
        const int nb = wave * 16;
#pragma unroll
        for (int l = 0; l < 3; ++l) {
            const _Float16* cur = (l & 1) ? bufB : bufA;
            _Float16*       nxt = (l & 1) ? bufA : bufB;
            const _Float16* wl  = wmid + (e * 3 + l) * (4 * 2 * 128 * 16);
            v16h b[4];
#pragma unroll
            for (int kb = 0; kb < 4; ++kb)
                b[kb] = load_b_pack(wl + kb * (2 * 128 * 16), 128, nb, lane);
            float bias = ebm[(e * 3 + l) * 128 + nb + (lane & 15)];
#pragma unroll
            for (int mi = 0; mi < 4; ++mi) {
                int mb = mi * 16;
                v8f acc = {};
#pragma unroll
                for (int kb = 0; kb < 4; ++kb) {
                    v16h a = load_a_lds(cur, 128, mb, kb * 32, lane);
                    acc = wmma_f16(a, b[kb], acc);
                }
                store_act(nxt, 128, mb, nb, acc, bias, lane);
            }
            __syncthreads();
        }

        // Output projection H->1 (+bias), gate-weighted accumulate.
        // After 3 layers the activations are in bufB. All 256 threads active:
        // 32 MACs each (v8h chunks), then 2-step cross-lane reduce.
        {
            const _Float16* hrow = bufB + tok * 128 + part * 32;
            const float*    w    = ewo + e * 128 + part * 32;
            float s = 0.0f;
#pragma unroll
            for (int c = 0; c < 4; ++c) {
                v8h hv = *(const v8h*)(hrow + c * 8);
#pragma unroll
                for (int j = 0; j < 8; ++j)
                    s += (float)hv[j] * w[c * 8 + j];
            }
            s += __shfl_xor(s, 1, 32);
            s += __shfl_xor(s, 2, 32);
            if (part == 0) yacc += gates[tok * 16 + e] * (s + ebo[e]);
        }
        __syncthreads();
    }

    if (part == 0) out[n0 + tok] = yacc;
}

// ---------------------------------------------------------------------------
extern "C" void kernel_launch(void* const* d_in, const int* in_sizes, int n_in,
                              void* d_out, int out_size, void* d_ws, size_t ws_size,
                              hipStream_t stream) {
    (void)in_sizes; (void)n_in; (void)out_size; (void)ws_size;
    const float* x   = (const float*)d_in[0];
    const float* ew1 = (const float*)d_in[1];
    const float* eb1 = (const float*)d_in[2];
    const float* ewm = (const float*)d_in[3];
    const float* ebm = (const float*)d_in[4];
    const float* ewo = (const float*)d_in[5];
    const float* ebo = (const float*)d_in[6];
    const float* gw1 = (const float*)d_in[7];
    const float* gb1 = (const float*)d_in[8];
    const float* gwm = (const float*)d_in[9];
    const float* gbm = (const float*)d_in[10];
    const float* gwo = (const float*)d_in[11];
    const float* gbo = (const float*)d_in[12];

    _Float16* ws = (_Float16*)d_ws;

    const int pack_total = EWM_PACK + GWM_PACK + GWO_PACK;
    MoE_pack_kernel<<<(pack_total + 255) / 256, 256, 0, stream>>>(ewm, gwm, gwo, ws);

    MoE_main_kernel<<<65536 / 64, 256, 0, stream>>>(
        x, ew1, eb1, ebm, ewo, ebo, gw1, gb1, gbm, gbo,
        ws, ws + EWM_PACK, ws + EWM_PACK + GWM_PACK, (float*)d_out);
}
